// MixtureOfExpertFFN_5909874999573
// MI455X (gfx1250) — matrix operand order repaired and checked by reading, not compiled, not run on hardware
//
#include <hip/hip_runtime.h>
#include <hip/hip_bf16.h>

typedef __bf16 v16bf __attribute__((ext_vector_type(16)));
typedef float  v8f   __attribute__((ext_vector_type(8)));
typedef unsigned int vu4 __attribute__((ext_vector_type(4)));

union BfFrag { v16bf v; vu4 q[2]; };

__device__ __forceinline__ __bf16 f2bf(float f) {
    unsigned u = __float_as_uint(f);
    unsigned r = (u + 0x7FFFu + ((u >> 16) & 1u)) >> 16;
    unsigned short s = (unsigned short)r;
    return __builtin_bit_cast(__bf16, s);
}

__device__ __forceinline__ float gelu_tanh(float x) {
    const float k0 = 0.7978845608028654f;   // sqrt(2/pi)
    const float k1 = 0.044715f;
    float x3 = x * x * x;
    return 0.5f * x * (1.0f + tanhf(k0 * (x + k1 * x3)));
}

// gfx1250 async global->LDS copy (ASYNCcnt-tracked). lds_off = LDS byte offset
// (low 32 bits of a flat __shared__ pointer), gaddr = 64-bit global address.
__device__ __forceinline__ void async_copy_b128(unsigned lds_off, unsigned long long gaddr) {
    asm volatile("global_load_async_to_lds_b128 %0, %1, off"
                 :: "v"(lds_off), "v"(gaddr) : "memory");
}
__device__ __forceinline__ void wait_async0() {
    asm volatile("s_wait_asynccnt 0x0" ::: "memory");
}

// ---------------------------------------------------------------------------
// 1) Gating: logits = x @ Wg, softmax, top-2, renormalize, count per expert
// ---------------------------------------------------------------------------
__global__ __launch_bounds__(256) void moe_gate_kernel(
    const float* __restrict__ x, const float* __restrict__ Wg,
    int* __restrict__ ee, float* __restrict__ w0a, float* __restrict__ w1a,
    int* __restrict__ counts, int T, int H)
{
    __shared__ float sWg[1024 * 8];          // 32 KB, whole gate matrix
    for (int i = threadIdx.x; i < H * 8; i += 256) sWg[i] = Wg[i];
    __syncthreads();

    int t = blockIdx.x * 256 + threadIdx.x;
    if (t >= T) return;

    const float* xr = x + (size_t)t * H;
    float acc[8] = {0.f, 0.f, 0.f, 0.f, 0.f, 0.f, 0.f, 0.f};
    for (int h = 0; h < H; h += 4) {
        float4 xv = *(const float4*)(xr + h);
        const float* w0 = &sWg[(h + 0) * 8];
        const float* w1 = &sWg[(h + 1) * 8];
        const float* w2 = &sWg[(h + 2) * 8];
        const float* w3 = &sWg[(h + 3) * 8];
        #pragma unroll
        for (int e = 0; e < 8; ++e)
            acc[e] += xv.x * w0[e] + xv.y * w1[e] + xv.z * w2[e] + xv.w * w3[e];
    }
    float mx = acc[0];
    #pragma unroll
    for (int e = 1; e < 8; ++e) mx = fmaxf(mx, acc[e]);
    float p[8];
    #pragma unroll
    for (int e = 0; e < 8; ++e) p[e] = __expf(acc[e] - mx);

    int i0 = 0; float p0 = p[0];
    #pragma unroll
    for (int e = 1; e < 8; ++e) if (p[e] > p0) { p0 = p[e]; i0 = e; }
    int i1 = -1; float p1 = -1.f;
    #pragma unroll
    for (int e = 0; e < 8; ++e) if (e != i0 && p[e] > p1) { p1 = p[e]; i1 = e; }

    float inv = 1.f / (p0 + p1);
    ee[t]  = i0 | (i1 << 8);
    w0a[t] = p0 * inv;
    w1a[t] = p1 * inv;
    atomicAdd(&counts[i0], 1);
    atomicAdd(&counts[i1], 1);
}

// ---------------------------------------------------------------------------
// 2) Exclusive prefix over expert counts (E=8, single thread)
// ---------------------------------------------------------------------------
__global__ void moe_scan_kernel(const int* __restrict__ counts,
                                int* __restrict__ offs, int* __restrict__ cursor, int E)
{
    if (threadIdx.x == 0 && blockIdx.x == 0) {
        int o = 0;
        for (int e = 0; e < E; ++e) { offs[e] = o; cursor[e] = o; o += counts[e]; }
        offs[E] = o;
    }
}

// ---------------------------------------------------------------------------
// 3) Scatter tokens into per-expert contiguous row ranges
// ---------------------------------------------------------------------------
__global__ __launch_bounds__(256) void moe_scatter_kernel(
    const int* __restrict__ ee, const float* __restrict__ w0a, const float* __restrict__ w1a,
    int* __restrict__ cursor, int* __restrict__ tok_idx, float* __restrict__ tok_w, int T)
{
    int t = blockIdx.x * 256 + threadIdx.x;
    if (t >= T) return;
    int pk = ee[t];
    int e0 = pk & 255, e1 = (pk >> 8) & 255;
    int r0 = atomicAdd(&cursor[e0], 1);
    tok_idx[r0] = t; tok_w[r0] = w0a[t];
    int r1 = atomicAdd(&cursor[e1], 1);
    tok_idx[r1] = t; tok_w[r1] = w1a[t];
}

// ---------------------------------------------------------------------------
// 4) fp32 -> bf16 conversions (weights transposed to N-major for WMMA B loads)
// ---------------------------------------------------------------------------
__global__ __launch_bounds__(256) void moe_cvt_x_kernel(
    const float* __restrict__ src, __bf16* __restrict__ dst, size_t n)
{
    size_t i = (size_t)blockIdx.x * 256 + threadIdx.x;
    size_t stride = (size_t)gridDim.x * 256;
    for (; i < n; i += stride) dst[i] = f2bf(src[i]);
}

// W1: [E][H][F] fp32  ->  w1t: [E][F][H] bf16   (column of N contiguous over K)
__global__ __launch_bounds__(256) void moe_cvt_w1_kernel(
    const float* __restrict__ W1, __bf16* __restrict__ w1t, int H, int F, size_t n)
{
    size_t i = (size_t)blockIdx.x * 256 + threadIdx.x;
    size_t stride = (size_t)gridDim.x * 256;
    for (; i < n; i += stride) {
        int f = (int)(i % F);
        size_t r = i / F;
        int h = (int)(r % H);
        int e = (int)(r / H);
        w1t[((size_t)e * F + f) * H + h] = f2bf(W1[i]);
    }
}

// W2: [E][F][H] fp32  ->  w2t: [E][H][F] bf16
__global__ __launch_bounds__(256) void moe_cvt_w2_kernel(
    const float* __restrict__ W2, __bf16* __restrict__ w2t, int H, int F, size_t n)
{
    size_t i = (size_t)blockIdx.x * 256 + threadIdx.x;
    size_t stride = (size_t)gridDim.x * 256;
    for (; i < n; i += stride) {
        int h = (int)(i % H);
        size_t r = i / H;
        int f = (int)(r % F);
        int e = (int)(r / F);
        w2t[((size_t)e * H + h) * F + f] = f2bf(W2[i]);
    }
}

// ---------------------------------------------------------------------------
// Grouped WMMA GEMMs.
//   block = 8 waves, block tile M=32 x N=256, K chunked (KC=256) through LDS
//   wave tile  = 32x32 -> 2x2 grid of 16x16 WMMA accumulators
//   A tile staged by double-buffered global_load_async_to_lds_b128 (gather,
//   ragged rows clamped to the last valid row; bogus C rows dropped in the
//   epilogue so no zero-fill is needed)
// ---------------------------------------------------------------------------
#define MOE_KC 256
#define MOE_MR 32

// stage one 32 x KC bf16 tile: 1024 b128 transfers over 256 threads
__device__ __forceinline__ void stage_tile_async(
    const __bf16* __restrict__ base, const int* __restrict__ row_ptr, // row_ptr: token ids or null
    int row0, int nrows, size_t ld, int k0, const __bf16* lds_base, int tid)
{
    #pragma unroll
    for (int j = 0; j < 4; ++j) {
        int i  = tid + j * 256;                  // 0..1023
        int r  = i >> 5;                         // row 0..31
        int c8 = i & 31;                         // 8-elem column chunk
        int rr = min(r, nrows - 1);
        size_t src_row = row_ptr ? (size_t)row_ptr[row0 + rr] : (size_t)(row0 + rr);
        unsigned long long g = (unsigned long long)(const void*)(base + src_row * ld + k0 + c8 * 8);
        unsigned lds = (unsigned)(unsigned long long)(const void*)(lds_base + r * MOE_KC + c8 * 8);
        async_copy_b128(lds, g);
    }
}

#define WMMA_BF16(A, B, C) \
    __builtin_amdgcn_wmma_f32_16x16x32_bf16(false, (A), false, (B), (short)0, (C), false, false)

// ---------------------------------------------------------------------------
// 5) GEMM 1: h = gelu(gather(x) @ W1[e] + b1[e]), bf16 out
// ---------------------------------------------------------------------------
__global__ __launch_bounds__(256) void moe_gemm1_kernel(
    const __bf16* __restrict__ xbf, const __bf16* __restrict__ w1t,
    const float* __restrict__ b1, const int* __restrict__ offs,
    const int* __restrict__ tok_idx, __bf16* __restrict__ hbuf,
    int H, int F)
{
    const int e    = blockIdx.z;
    const int rbeg = offs[e], rend = offs[e + 1];
    const int row0 = rbeg + blockIdx.y * MOE_MR;
    if (row0 >= rend) return;
    const int nrows = min(MOE_MR, rend - row0);
    const int n0   = blockIdx.x * 256;
    const int lane = threadIdx.x & 31;
    const int wv   = threadIdx.x >> 5;

    __shared__ __bf16 As[2 * MOE_MR * MOE_KC];   // 32 KB double buffer

    v8f c00 = {}, c01 = {}, c10 = {}, c11 = {};
    const int na = n0 + wv * 32 + (lane & 15);   // first N column of this lane
    const __bf16* wb0 = w1t + ((size_t)e * F + na) * (size_t)H;
    const __bf16* wb1 = wb0 + (size_t)16 * H;    // +16 columns

    stage_tile_async(xbf, tok_idx, row0, nrows, (size_t)H, 0, As, threadIdx.x);

    for (int k0 = 0, buf = 0; k0 < H; k0 += MOE_KC, buf ^= 1) {
        wait_async0();
        __syncthreads();
        if (k0 + MOE_KC < H)
            stage_tile_async(xbf, tok_idx, row0, nrows, (size_t)H, k0 + MOE_KC,
                             As + (buf ^ 1) * MOE_MR * MOE_KC, threadIdx.x);
        __builtin_prefetch(wb0 + k0 + MOE_KC, 0, 1);
        const __bf16* Asb = As + buf * MOE_MR * MOE_KC;

        #pragma unroll
        for (int kk = 0; kk < MOE_KC; kk += 32) {
            BfFrag a0, a1, b0, b1f;
            const int ml = lane & 15;
            const int kb = kk + ((lane >> 4) << 3);       // A frag: +8 K for hi half-wave
            a0.q[0] = *(const vu4*)&Asb[ml * MOE_KC + kb];
            a0.q[1] = *(const vu4*)&Asb[ml * MOE_KC + kb + 16];
            a1.q[0] = *(const vu4*)&Asb[(ml + 16) * MOE_KC + kb];
            a1.q[1] = *(const vu4*)&Asb[(ml + 16) * MOE_KC + kb + 16];
            const int kg = k0 + kk + ((lane >> 4) << 3);  // B frag: +8 K for hi half-wave
            b0.q[0]  = *(const vu4*)(wb0 + kg);
            b0.q[1]  = *(const vu4*)(wb0 + kg + 16);
            b1f.q[0] = *(const vu4*)(wb1 + kg);
            b1f.q[1] = *(const vu4*)(wb1 + kg + 16);
            c00 = WMMA_BF16(a0.v, b0.v,  c00);
            c01 = WMMA_BF16(a0.v, b1f.v, c01);
            c10 = WMMA_BF16(a1.v, b0.v,  c10);
            c11 = WMMA_BF16(a1.v, b1f.v, c11);
        }
    }

    const float bias0 = b1[(size_t)e * F + na];
    const float bias1 = b1[(size_t)e * F + na + 16];
    const v8f* cs[4] = { &c00, &c01, &c10, &c11 };
    #pragma unroll
    for (int tm = 0; tm < 2; ++tm) {
        #pragma unroll
        for (int tn = 0; tn < 2; ++tn) {
            const v8f& c = *cs[tm * 2 + tn];
            const int n = na + tn * 16;
            const float bias = tn ? bias1 : bias0;
            #pragma unroll
            for (int r = 0; r < 8; ++r) {
                int m = tm * 16 + r + ((lane >> 4) << 3);
                if (m < nrows) {
                    float g = gelu_tanh(c[r] + bias);
                    hbuf[(size_t)(row0 + m) * F + n] = f2bf(g);
                }
            }
        }
    }
}

// ---------------------------------------------------------------------------
// 6) GEMM 2: out[t] += w * (h @ W2[e] + b2[e]), fp32 atomic combine
// ---------------------------------------------------------------------------
__global__ __launch_bounds__(256) void moe_gemm2_kernel(
    const __bf16* __restrict__ hbuf, const __bf16* __restrict__ w2t,
    const float* __restrict__ b2, const int* __restrict__ offs,
    const int* __restrict__ tok_idx, const float* __restrict__ tok_w,
    float* __restrict__ out, int H, int F)
{
    const int e    = blockIdx.z;
    const int rbeg = offs[e], rend = offs[e + 1];
    const int row0 = rbeg + blockIdx.y * MOE_MR;
    if (row0 >= rend) return;
    const int nrows = min(MOE_MR, rend - row0);
    const int n0   = blockIdx.x * 256;
    const int lane = threadIdx.x & 31;
    const int wv   = threadIdx.x >> 5;

    __shared__ __bf16 As[2 * MOE_MR * MOE_KC];

    v8f c00 = {}, c01 = {}, c10 = {}, c11 = {};
    const int na = n0 + wv * 32 + (lane & 15);
    const __bf16* wb0 = w2t + ((size_t)e * H + na) * (size_t)F;
    const __bf16* wb1 = wb0 + (size_t)16 * F;

    // A rows are contiguous assignment rows of hbuf (no gather)
    stage_tile_async(hbuf, nullptr, row0, nrows, (size_t)F, 0, As, threadIdx.x);

    for (int k0 = 0, buf = 0; k0 < F; k0 += MOE_KC, buf ^= 1) {
        wait_async0();
        __syncthreads();
        if (k0 + MOE_KC < F)
            stage_tile_async(hbuf, nullptr, row0, nrows, (size_t)F, k0 + MOE_KC,
                             As + (buf ^ 1) * MOE_MR * MOE_KC, threadIdx.x);
        __builtin_prefetch(wb0 + k0 + MOE_KC, 0, 1);
        const __bf16* Asb = As + buf * MOE_MR * MOE_KC;

        #pragma unroll
        for (int kk = 0; kk < MOE_KC; kk += 32) {
            BfFrag a0, a1, b0, b1f;
            const int ml = lane & 15;
            const int kb = kk + ((lane >> 4) << 3);
            a0.q[0] = *(const vu4*)&Asb[ml * MOE_KC + kb];
            a0.q[1] = *(const vu4*)&Asb[ml * MOE_KC + kb + 16];
            a1.q[0] = *(const vu4*)&Asb[(ml + 16) * MOE_KC + kb];
            a1.q[1] = *(const vu4*)&Asb[(ml + 16) * MOE_KC + kb + 16];
            const int kg = k0 + kk + ((lane >> 4) << 3);
            b0.q[0]  = *(const vu4*)(wb0 + kg);
            b0.q[1]  = *(const vu4*)(wb0 + kg + 16);
            b1f.q[0] = *(const vu4*)(wb1 + kg);
            b1f.q[1] = *(const vu4*)(wb1 + kg + 16);
            c00 = WMMA_BF16(a0.v, b0.v,  c00);
            c01 = WMMA_BF16(a0.v, b1f.v, c01);
            c10 = WMMA_BF16(a1.v, b0.v,  c10);
            c11 = WMMA_BF16(a1.v, b1f.v, c11);
        }
    }

    const float bias0 = b2[(size_t)e * H + na];
    const float bias1 = b2[(size_t)e * H + na + 16];
    const v8f* cs[4] = { &c00, &c01, &c10, &c11 };
    #pragma unroll
    for (int tm = 0; tm < 2; ++tm) {
        #pragma unroll
        for (int tn = 0; tn < 2; ++tn) {
            const v8f& c = *cs[tm * 2 + tn];
            const int n = na + tn * 16;
            const float bias = tn ? bias1 : bias0;
            #pragma unroll
            for (int r = 0; r < 8; ++r) {
                int m = tm * 16 + r + ((lane >> 4) << 3);
                if (m < nrows) {
                    int row = row0 + m;
                    int t = tok_idx[row];
                    float w = tok_w[row];
                    atomicAdd(&out[(size_t)t * H + n], (c[r] + bias) * w);
                }
            }
        }
    }
}

// ---------------------------------------------------------------------------
// Host launcher
// ---------------------------------------------------------------------------
extern "C" void kernel_launch(void* const* d_in, const int* in_sizes, int n_in,
                              void* d_out, int out_size, void* d_ws, size_t ws_size,
                              hipStream_t stream) {
    const int B = 4, S = 2048, H = 1024, F = 4096, E = 8;
    const int T = B * S;                 // 8192 tokens
    const int A = 2 * T;                 // total routed assignments (top-2)

    const float* x  = (const float*)d_in[0];
    const float* Wg = (const float*)d_in[1];
    const float* W1 = (const float*)d_in[2];
    const float* b1 = (const float*)d_in[3];
    const float* W2 = (const float*)d_in[4];
    const float* b2 = (const float*)d_in[5];
    float* out = (float*)d_out;

    char* ws = (char*)d_ws;
    size_t off = 0;
    auto alloc = [&](size_t bytes) {
        size_t o = off;
        off = (off + bytes + 255) & ~(size_t)255;
        return o;
    };
    __bf16* xbf     = (__bf16*)(ws + alloc((size_t)T * H * 2));
    __bf16* w1t     = (__bf16*)(ws + alloc((size_t)E * H * F * 2));
    __bf16* w2t     = (__bf16*)(ws + alloc((size_t)E * F * H * 2));
    __bf16* hbuf    = (__bf16*)(ws + alloc((size_t)A * F * 2));
    int*    tok_idx = (int*)   (ws + alloc((size_t)A * 4));
    float*  tok_w   = (float*) (ws + alloc((size_t)A * 4));
    int*    ee      = (int*)   (ws + alloc((size_t)T * 4));
    float*  w0a     = (float*) (ws + alloc((size_t)T * 4));
    float*  w1a     = (float*) (ws + alloc((size_t)T * 4));
    int*    counts  = (int*)   (ws + alloc((size_t)E * 4));
    int*    offs    = (int*)   (ws + alloc((size_t)(E + 1) * 4));
    int*    cursor  = (int*)   (ws + alloc((size_t)E * 4));

    hipMemsetAsync(out,    0, (size_t)T * H * sizeof(float), stream);
    hipMemsetAsync(counts, 0, (size_t)E * sizeof(int),       stream);

    // routing
    moe_gate_kernel<<<T / 256, 256, 0, stream>>>(x, Wg, ee, w0a, w1a, counts, T, H);
    moe_scan_kernel<<<1, 32, 0, stream>>>(counts, offs, cursor, E);
    moe_scatter_kernel<<<T / 256, 256, 0, stream>>>(ee, w0a, w1a, cursor, tok_idx, tok_w, T);

    // bf16 conversion (weights become L2-resident at 134 MB total)
    moe_cvt_x_kernel<<<2048, 256, 0, stream>>>(x, xbf, (size_t)T * H);
    moe_cvt_w1_kernel<<<4096, 256, 0, stream>>>(W1, w1t, H, F, (size_t)E * H * F);
    moe_cvt_w2_kernel<<<4096, 256, 0, stream>>>(W2, w2t, H, F, (size_t)E * F * H);

    // grouped expert GEMMs (32-row M tiles; worst case one expert owns all T rows)
    dim3 g1(F / 256, T / MOE_MR, E);   // (16, 256, 8)
    moe_gemm1_kernel<<<g1, 256, 0, stream>>>(xbf, w1t, b1, offs, tok_idx, hbuf, H, F);
    dim3 g2(H / 256, T / MOE_MR, E);   // (4, 256, 8)
    moe_gemm2_kernel<<<g2, 256, 0, stream>>>(hbuf, w2t, b2, offs, tok_idx, tok_w, out, H, F);
}
